// BattleTransformer_31035433681288
// MI455X (gfx1250) — matrix operand order repaired and checked by reading, not compiled, not run on hardware
//
#include <hip/hip_runtime.h>

typedef __attribute__((ext_vector_type(16))) _Float16 v16h;
typedef __attribute__((ext_vector_type(8)))  float    v8f;

#define HN    187
#define SREAL 189
#define SP    192
#define DM    128
#define NHEAD 4
#define HDIM  32
#define NLAY  4
#define FFD   512
#define MSN   20
#define NCTN  256
#define CEN   16
#define BN    1024
#define NW    8     // waves per block (256 threads, wave32)

// f16 weight workspace layout (halves, per layer)
#define WL_QKV 0
#define WL_O   49152
#define WL_FF1 65536
#define WL_FF2 131072
#define WL_TOT 196608

// LDS layout (byte offsets)
#define OFF_X    0        // f32 [192][128]   98304
#define OFF_A    98304    // f16 [192][128]   49152
#define OFF_O    147456   // f16 [192][128]   49152
#define OFF_SCR  196608   // 86016 B scratch (QKV 36864 + P 49152 | FF hidden | preproc staging | head scratch)
#define OFF_EXT  282624   // gvec/pooled/valid/pos
#define SMEM_BYTES 283904

struct Params {
  const float *scalars, *stacks, *obstacles, *reach;
  const int   *n_stacks;
  const float *creature_emb, *hex_pos_emb, *tok_type_emb;
  const float *hex_proj_w, *hex_proj_b, *hex_ln_g, *hex_ln_b;
  const float *hero_proj_w, *hero_proj_b, *hero_ln_g, *hero_ln_b;
  const float *global_w, *global_b;
  const float *ln1_g, *ln1_b, *qkv_b, *o_b, *ln2_g, *ln2_b, *ff1_b, *ff2_b;
  const float *post_g, *post_b;
  const float *at_w1, *at_b1, *at_w2, *at_b2;
  const float *hx_w1, *hx_b1, *hx_w2, *hx_b2;
  const float *tg_w1, *tg_b1, *tg_w2, *tg_b2;
  const float *vl_w1, *vl_b1, *vl_w2, *vl_b2;
  const _Float16 *wh;  // converted f16 weights in d_ws
  float *out;
};

// ---------------------------------------------------------------- utilities
__device__ inline float wred(float v) {
#pragma unroll
  for (int o = 16; o > 0; o >>= 1) v += __shfl_xor(v, o, 32);
  return v;
}

// A-fragment (16x32, f16): row M = lane%16, k(i) = (i/8)*16 + 8*(lane/16) + i%8
__device__ inline v16h ld_fragA(const _Float16* base, int stride) {
  int lane = threadIdx.x & 31;
  const _Float16* p = base + (lane & 15) * stride + ((lane >> 4) << 3);
  v16h f;
#pragma unroll
  for (int i = 0; i < 8; ++i) { f[i] = p[i]; f[i + 8] = p[16 + i]; }
  return f;
}
// B-fragment (32x16, f16) gathered from a row-major matrix whose rows are the
// N dimension (i.e. B[k][n] = M[n][k]): row = lane%16, k(i) = (i/8)*16 + 8*(lane/16) + i%8
__device__ inline v16h ld_fragB(const _Float16* base, int stride) {
  int lane = threadIdx.x & 31;
  const _Float16* p = base + (lane & 15) * stride + ((lane >> 4) << 3);
  v16h f;
#pragma unroll
  for (int i = 0; i < 8; ++i) { f[i] = p[i]; f[i + 8] = p[16 + i]; }
  return f;
}
__device__ inline v8f wmma16(v16h a, v16h b, v8f c) {
  return __builtin_amdgcn_wmma_f32_16x16x32_f16(false, a, false, b, (short)0, c, false, false);
}

// LayerNorm rows of fp32 LDS -> f16 LDS (one warp per row)
__device__ inline void ln_f16(const float* Xb, _Float16* Ab, const float* g,
                              const float* bb, int rows, int lane, int wid) {
  for (int s = wid; s < rows; s += NW) {
    const float* xr = Xb + s * DM;
    float v[4];
#pragma unroll
    for (int q = 0; q < 4; ++q) v[q] = xr[lane + 32 * q];
    float mean = wred(v[0] + v[1] + v[2] + v[3]) * (1.f / DM);
    float var = wred((v[0]-mean)*(v[0]-mean) + (v[1]-mean)*(v[1]-mean) +
                     (v[2]-mean)*(v[2]-mean) + (v[3]-mean)*(v[3]-mean)) * (1.f / DM);
    float inv = rsqrtf(var + 1e-5f);
    _Float16* ar = Ab + s * DM;
#pragma unroll
    for (int q = 0; q < 4; ++q) {
      int d = lane + 32 * q;
      ar[d] = (_Float16)((v[q] - mean) * inv * g[d] + bb[d]);
    }
  }
}

// ---------------------------------------------------------------- weight cvt
__global__ void convert_weights(const float* qkv, const float* ow,
                                const float* ff1, const float* ff2, _Float16* dst) {
  int idx = blockIdx.x * 256 + threadIdx.x;
  int total = NLAY * WL_TOT;
  if (idx >= total) return;
  int l = idx / WL_TOT, r = idx % WL_TOT;
  float v;
  if (r < WL_O)        v = qkv[l * (3 * DM * DM) + r];
  else if (r < WL_FF1) v = ow [l * (DM * DM)     + (r - WL_O)];
  else if (r < WL_FF2) v = ff1[l * (FFD * DM)    + (r - WL_FF1)];
  else                 v = ff2[l * (DM * FFD)    + (r - WL_FF2)];
  dst[idx] = (_Float16)v;
}

// ---------------------------------------------------------------- fused fwd
__global__ __launch_bounds__(256, 1) void battle_fused(Params p) {
  extern __shared__ char smem[];
  float*    X    = (float*)(smem + OFF_X);      // [192][128] residual stream
  _Float16* Abuf = (_Float16*)(smem + OFF_A);   // [192][128] LN'd activations
  _Float16* Obuf = (_Float16*)(smem + OFF_O);   // [192][128] attention output
  float*    gvec   = (float*)(smem + OFF_EXT);  // [128]
  float*    pooled = gvec + DM;                 // [128]
  float*    validf = pooled + DM;               // [20]
  int*      posi   = (int*)(validf + MSN);      // [20]

  const int b = blockIdx.x;
  const int tid = threadIdx.x;
  const int lane = tid & 31, wid = tid >> 5;
  const int nfrag = lane & 15, mbase = (lane >> 4) << 3;

  // ================= preprocessing: build tokens in X =================
  {
    float* hf   = (float*)(smem + OFF_SCR);     // [187][45] staged hex features
    int*   cidb = (int*)(smem + OFF_SCR + HN * 45 * 4);
    for (int i = tid; i < HN * 45; i += 256) hf[i] = 0.f;
    for (int i = tid; i < HN; i += 256) cidb[i] = 0;
    __syncthreads();

    const float* sc = p.scalars + b * 19;
    if (tid < MSN) {
      const float* st = p.stacks + (size_t)(b * MSN + tid) * 35;
      int pos = (int)st[18];
      float alive = st[23];
      int ns = p.n_stacks[b];
      bool valid = (pos >= 0) && (pos < HN) && (alive >= 0.5f) && (tid < ns);
      posi[tid] = pos;
      validf[tid] = valid ? 1.f : 0.f;
      if (valid) {
        float* f = hf + pos * 45;
        float max_hp = fmaxf(st[4], 1.f);
        f[0] = 1.f; f[1] = st[2] * 0.001f; f[2] = st[3] / max_hp;
#pragma unroll
        for (int c = 0; c < 8; ++c) f[3 + c] = st[8 + c] * 0.01f;
        f[11] = st[16] * 0.05f; f[12] = st[17] * 0.05f;
        f[13] = st[20];
        f[14] = (st[20] == sc[3]) ? 1.f : 0.f;
        f[15] = alive;
#pragma unroll
        for (int c = 0; c < 7; ++c) f[16 + c] = st[24 + c];
        f[23] = st[31] * (1.f / 30.f); f[24] = st[33] * 0.2f; f[25] = st[34] * 0.1f;
        f[26] = (st[0] == sc[2]) ? 1.f : 0.f;
        int cv = (int)st[1]; if (cv > NCTN - 1) cv = NCTN - 1;
        cidb[pos] = cv;
      }
    }
    __syncthreads();
    for (int i = tid; i < HN; i += 256) hf[i * 45 + 27] = p.reach[b * HN + i];
    if (tid < 10) {
      const float* ob = p.obstacles + (size_t)(b * 10 + tid) * 3;
      int op = (int)ob[2];
      if (ob[0] > 0.f && op >= 0 && op < HN) hf[op * 45 + 28] = 1.f;
    }
    __syncthreads();
    // hex projection (45 -> 128), pre-LN values into X
    for (int i = tid; i < HN * DM; i += 256) {
      int s = i >> 7, d = i & 127;
      const float* w = p.hex_proj_w + d * 45;
      const float* f = hf + s * 45;
      float acc = p.hex_proj_b[d];
#pragma unroll
      for (int c = 0; c < 29; ++c) acc += f[c] * w[c];
      const float* ce = p.creature_emb + cidb[s] * CEN;
#pragma unroll
      for (int c = 0; c < CEN; ++c) acc += ce[c] * w[29 + c];
      X[s * DM + d] = acc;
    }
    if (tid < DM) {
      int d = tid;
      const float* w = p.hero_proj_w + d * 5;
      float a0 = sc[8], a1 = sc[11] / 300.f, a2 = sc[10] * 0.1f, a3 = sc[12] * 0.1f;
      float f0 = sc[14], f1 = sc[17] / 300.f, f2 = sc[16] * 0.1f, f3 = sc[18] * 0.1f;
      X[187 * DM + d] = p.hero_proj_b[d] + a0*w[0] + a1*w[1] + a2*w[2] + a3*w[3];
      X[188 * DM + d] = p.hero_proj_b[d] + f0*w[0] + f1*w[1] + f2*w[2] + f3*w[3] + w[4];
      const float* gw = p.global_w + d * 5;
      gvec[d] = p.global_b[d] + (sc[1] * 0.02f) * gw[0] + (sc[4] * 0.1f) * gw[1] +
                (sc[5] * 0.1f) * gw[2] + sc[6] * gw[3] + sc[3] * gw[4];
    }
    for (int i = tid; i < 3 * DM; i += 256) X[SREAL * DM + i] = 0.f;  // pad rows
    __syncthreads();
    // token LN + pos/type embeddings + global broadcast
    for (int s = wid; s < SREAL; s += NW) {
      float* xr = X + s * DM;
      float v[4];
#pragma unroll
      for (int q = 0; q < 4; ++q) v[q] = xr[lane + 32 * q];
      float mean = wred(v[0] + v[1] + v[2] + v[3]) * (1.f / DM);
      float var = wred((v[0]-mean)*(v[0]-mean) + (v[1]-mean)*(v[1]-mean) +
                       (v[2]-mean)*(v[2]-mean) + (v[3]-mean)*(v[3]-mean)) * (1.f / DM);
      float inv = rsqrtf(var + 1e-5f);
      const float* g  = (s < HN) ? p.hex_ln_g : p.hero_ln_g;
      const float* bb = (s < HN) ? p.hex_ln_b : p.hero_ln_b;
      int tt = (s < HN) ? 0 : (s - HN + 1);
#pragma unroll
      for (int q = 0; q < 4; ++q) {
        int d = lane + 32 * q;
        float y = (v[q] - mean) * inv * g[d] + bb[d] + p.tok_type_emb[tt * DM + d] + gvec[d];
        if (s < HN) y += p.hex_pos_emb[s * DM + d];
        xr[d] = y;
      }
    }
    __syncthreads();
  }

  // ================= transformer layers =================
  _Float16* Qs = (_Float16*)(smem + OFF_SCR);          // [192][32]; Ks follows at +SP*HDIM
  _Float16* Vt = Qs + 2 * SP * HDIM;                   // [32][192] (transposed)
  _Float16* Pb = (_Float16*)(smem + OFF_SCR + 36864);  // [8 waves][16][192]
  _Float16* Hb = (_Float16*)(smem + OFF_SCR);          // [16][512] (MLP phase)

  for (int l = 0; l < NLAY; ++l) {
    const _Float16* Wl   = p.wh + (size_t)l * WL_TOT;
    const _Float16* Wqkv = Wl + WL_QKV;
    const _Float16* Wo   = Wl + WL_O;
    const _Float16* Wf1  = Wl + WL_FF1;
    const _Float16* Wf2  = Wl + WL_FF2;

    // ---- LN1 -> Abuf
    ln_f16(X, Abuf, p.ln1_g + l * DM, p.ln1_b + l * DM, SP, lane, wid);

    // ---- attention, one head at a time
    for (int h = 0; h < NHEAD; ++h) {
      __syncthreads();
      // Q,K for this head: 48 tiles (2 matrices x 12 mtiles x 2 ntiles)
      for (int t = wid; t < 48; t += NW) {
        int mi = t / 24, rem = t % 24, mt = rem >> 1, nt = rem & 1;
        int wrow0 = mi * DM + h * HDIM + nt * 16;
        const _Float16* Wb = Wqkv + (size_t)wrow0 * DM;
        v8f acc = {};
#pragma unroll
        for (int kc = 0; kc < 4; ++kc)
          acc = wmma16(ld_fragA(Abuf + mt * 16 * DM + kc * 32, DM),
                       ld_fragB(Wb + kc * 32, DM), acc);
        float bias = p.qkv_b[l * 3 * DM + wrow0 + nfrag];
        _Float16* dst = Qs + mi * (SP * HDIM) + (mt * 16 + mbase) * HDIM + nt * 16 + nfrag;
#pragma unroll
        for (int r = 0; r < 8; ++r)
          dst[r * HDIM] = (_Float16)(acc[r] + bias);
      }
      // V for this head: 24 tiles, stored transposed [32][192]
      for (int t = wid; t < 24; t += NW) {
        int mt = t >> 1, nt = t & 1;
        int wrow0 = 2 * DM + h * HDIM + nt * 16;
        const _Float16* Wb = Wqkv + (size_t)wrow0 * DM;
        v8f acc = {};
#pragma unroll
        for (int kc = 0; kc < 4; ++kc)
          acc = wmma16(ld_fragA(Abuf + mt * 16 * DM + kc * 32, DM),
                       ld_fragB(Wb + kc * 32, DM), acc);
        float bias = p.qkv_b[l * 3 * DM + wrow0 + nfrag];
        _Float16* dst = Vt + (nt * 16 + nfrag) * SP + mt * 16 + mbase;
#pragma unroll
        for (int r = 0; r < 8; ++r)
          dst[r] = (_Float16)(acc[r] + bias);
      }
      __syncthreads();
      // scores + softmax (registers) + P@V
      for (int qt = wid; qt < 12; qt += NW) {
        v16h aq = ld_fragA(Qs + qt * 16 * HDIM, HDIM);
        v8f sacc[12];
#pragma unroll
        for (int kt = 0; kt < 12; ++kt) {
          v8f z = {};
          sacc[kt] = wmma16(aq, ld_fragB(Qs + SP * HDIM + kt * 16 * HDIM, HDIM), z);
        }
        const float scale = 0.17677669529663687f;  // 1/sqrt(32)
        float rmax[8], rsum[8];
#pragma unroll
        for (int r = 0; r < 8; ++r) rmax[r] = -3.0e30f;
#pragma unroll
        for (int kt = 0; kt < 12; ++kt)
#pragma unroll
          for (int r = 0; r < 8; ++r) {
            float v = sacc[kt][r] * scale;
            if (kt == 11 && nfrag >= 13) v = -3.0e30f;  // mask pad keys 189..191
            sacc[kt][r] = v;
            rmax[r] = fmaxf(rmax[r], v);
          }
#pragma unroll
        for (int r = 0; r < 8; ++r) {
#pragma unroll
          for (int o = 1; o < 16; o <<= 1) rmax[r] = fmaxf(rmax[r], __shfl_xor(rmax[r], o, 32));
          rsum[r] = 0.f;
        }
#pragma unroll
        for (int kt = 0; kt < 12; ++kt)
#pragma unroll
          for (int r = 0; r < 8; ++r) {
            float e = __expf(sacc[kt][r] - rmax[r]);
            sacc[kt][r] = e; rsum[r] += e;
          }
#pragma unroll
        for (int r = 0; r < 8; ++r) {
#pragma unroll
          for (int o = 1; o < 16; o <<= 1) rsum[r] += __shfl_xor(rsum[r], o, 32);
          rsum[r] = 1.f / rsum[r];
        }
        _Float16* P = Pb + wid * 16 * SP;
#pragma unroll
        for (int kt = 0; kt < 12; ++kt)
#pragma unroll
          for (int r = 0; r < 8; ++r)
            P[(mbase + r) * SP + kt * 16 + nfrag] = (_Float16)(sacc[kt][r] * rsum[r]);
        // O = P @ V  (K = 192 = 6 chunks)
#pragma unroll
        for (int nt2 = 0; nt2 < 2; ++nt2) {
          v8f oacc = {};
#pragma unroll
          for (int kc = 0; kc < 6; ++kc)
            oacc = wmma16(ld_fragA(P + kc * 32, SP),
                          ld_fragB(Vt + nt2 * 16 * SP + kc * 32, SP), oacc);
          _Float16* dst = Obuf + (qt * 16 + mbase) * DM + h * HDIM + nt2 * 16 + nfrag;
#pragma unroll
          for (int r = 0; r < 8; ++r)
            dst[r * DM] = (_Float16)oacc[r];
        }
      }
    }
    __syncthreads();
    // ---- output projection + residual
    for (int t = wid; t < 96; t += NW) {
      int mt = t >> 3, nt = t & 7;
      const _Float16* Wb = Wo + (size_t)(nt * 16) * DM;
      v8f acc = {};
#pragma unroll
      for (int kc = 0; kc < 4; ++kc)
        acc = wmma16(ld_fragA(Obuf + mt * 16 * DM + kc * 32, DM),
                     ld_fragB(Wb + kc * 32, DM), acc);
      float bias = p.o_b[l * DM + nt * 16 + nfrag];
      float* dst = X + (mt * 16 + mbase) * DM + nt * 16 + nfrag;
#pragma unroll
      for (int r = 0; r < 8; ++r)
        dst[r * DM] += acc[r] + bias;
    }
    __syncthreads();
    // ---- LN2 -> Abuf
    ln_f16(X, Abuf, p.ln2_g + l * DM, p.ln2_b + l * DM, SP, lane, wid);
    __syncthreads();
    // ---- MLP, one 16-token tile at a time (hidden tile stays in LDS)
    for (int tt = 0; tt < 12; ++tt) {
      for (int t = wid; t < 32; t += NW) {
        const _Float16* Wb = Wf1 + (size_t)(t * 16) * DM;
        v8f acc = {};
#pragma unroll
        for (int kc = 0; kc < 4; ++kc)
          acc = wmma16(ld_fragA(Abuf + tt * 16 * DM + kc * 32, DM),
                       ld_fragB(Wb + kc * 32, DM), acc);
        float bias = p.ff1_b[l * FFD + t * 16 + nfrag];
        _Float16* dst = Hb + mbase * FFD + t * 16 + nfrag;
#pragma unroll
        for (int r = 0; r < 8; ++r)
          dst[r * FFD] = (_Float16)fmaxf(acc[r] + bias, 0.f);
      }
      __syncthreads();
      for (int t = wid; t < 8; t += NW) {
        const _Float16* Wb = Wf2 + (size_t)(t * 16) * FFD;
        v8f acc = {};
#pragma unroll
        for (int kc = 0; kc < 16; ++kc)
          acc = wmma16(ld_fragA(Hb + kc * 32, FFD), ld_fragB(Wb + kc * 32, FFD), acc);
        float bias = p.ff2_b[l * DM + t * 16 + nfrag];
        float* dst = X + (tt * 16 + mbase) * DM + t * 16 + nfrag;
#pragma unroll
        for (int r = 0; r < 8; ++r)
          dst[r * DM] += acc[r] + bias;
      }
      __syncthreads();
    }
  }

  // ================= post-LN (in place, real tokens) =================
  for (int s = wid; s < SREAL; s += NW) {
    float* xr = X + s * DM;
    float v[4];
#pragma unroll
    for (int q = 0; q < 4; ++q) v[q] = xr[lane + 32 * q];
    float mean = wred(v[0] + v[1] + v[2] + v[3]) * (1.f / DM);
    float var = wred((v[0]-mean)*(v[0]-mean) + (v[1]-mean)*(v[1]-mean) +
                     (v[2]-mean)*(v[2]-mean) + (v[3]-mean)*(v[3]-mean)) * (1.f / DM);
    float inv = rsqrtf(var + 1e-5f);
#pragma unroll
    for (int q = 0; q < 4; ++q) {
      int d = lane + 32 * q;
      xr[d] = (v[q] - mean) * inv * p.post_g[d] + p.post_b[d];
    }
  }
  __syncthreads();

  // ================= heads =================
  if (tid < DM) {
    float acc = 0.f;
    for (int s = 0; s < SREAL; ++s) acc += X[s * DM + tid];
    pooled[tid] = acc * (1.f / (float)SREAL);
  }
  __syncthreads();
  float* h1 = (float*)(smem + OFF_SCR);
  float* h2 = h1 + DM;
  if (tid < DM) {
    const float* w = p.at_w1 + tid * DM;
    float a = p.at_b1[tid];
    for (int k = 0; k < DM; ++k) a += pooled[k] * w[k];
    h1[tid] = fmaxf(a, 0.f);
  } else if (tid < 2 * DM) {
    int j = tid - DM;
    const float* w = p.vl_w1 + j * DM;
    float a = p.vl_b1[j];
    for (int k = 0; k < DM; ++k) a += pooled[k] * w[k];
    h2[j] = fmaxf(a, 0.f);
  }
  __syncthreads();
  if (tid < 5) {
    const float* w = p.at_w2 + tid * DM;
    float a = p.at_b2[tid];
    for (int k = 0; k < DM; ++k) a += h1[k] * w[k];
    p.out[b * 5 + tid] = a;
  }
  if (tid == 32) {
    float a = p.vl_b2[0];
    for (int k = 0; k < DM; ++k) a += h2[k] * p.vl_w2[k];
    p.out[BN * 5 + BN * HN + BN * MSN + b] = a;
  }
  // per-hex logits (warp per token)
  for (int s = wid; s < HN; s += NW) {
    const float* xs = X + s * DM;
    float part = 0.f;
#pragma unroll
    for (int jj = 0; jj < 2; ++jj) {
      int j = lane * 2 + jj;
      const float* w = p.hx_w1 + j * DM;
      float a = p.hx_b1[j];
      for (int k = 0; k < DM; ++k) a += xs[k] * w[k];
      part += fmaxf(a, 0.f) * p.hx_w2[j];
    }
    part = wred(part);
    if (lane == 0) p.out[BN * 5 + b * HN + s] = part + p.hx_b2[0];
  }
  // target logits (warp per stack)
  for (int j = wid; j < MSN; j += NW) {
    int pos = posi[j];
    if (pos < 0) pos = 0;
    if (pos > HN - 1) pos = HN - 1;
    const float* xs = X + pos * DM;
    float part = 0.f;
#pragma unroll
    for (int jj = 0; jj < 2; ++jj) {
      int q = lane * 2 + jj;
      const float* w = p.tg_w1 + q * DM;
      float a = p.tg_b1[q];
      for (int k = 0; k < DM; ++k) a += xs[k] * w[k];
      part += fmaxf(a, 0.f) * p.tg_w2[q];
    }
    part = wred(part);
    if (lane == 0) {
      float v = part + p.tg_b2[0];
      p.out[BN * 5 + BN * HN + b * MSN + j] = (validf[j] > 0.5f) ? v : -1.0e9f;
    }
  }
}

// ---------------------------------------------------------------- launcher
extern "C" void kernel_launch(void* const* d_in, const int* in_sizes, int n_in,
                              void* d_out, int out_size, void* d_ws, size_t ws_size,
                              hipStream_t stream) {
  Params p;
  p.scalars = (const float*)d_in[0];
  p.stacks = (const float*)d_in[1];
  p.obstacles = (const float*)d_in[2];
  p.reach = (const float*)d_in[3];
  p.n_stacks = (const int*)d_in[4];
  p.creature_emb = (const float*)d_in[5];
  p.hex_pos_emb = (const float*)d_in[6];
  p.tok_type_emb = (const float*)d_in[7];
  p.hex_proj_w = (const float*)d_in[8];
  p.hex_proj_b = (const float*)d_in[9];
  p.hex_ln_g = (const float*)d_in[10];
  p.hex_ln_b = (const float*)d_in[11];
  p.hero_proj_w = (const float*)d_in[12];
  p.hero_proj_b = (const float*)d_in[13];
  p.hero_ln_g = (const float*)d_in[14];
  p.hero_ln_b = (const float*)d_in[15];
  p.global_w = (const float*)d_in[16];
  p.global_b = (const float*)d_in[17];
  p.ln1_g = (const float*)d_in[18];
  p.ln1_b = (const float*)d_in[19];
  p.qkv_b = (const float*)d_in[21];
  p.o_b = (const float*)d_in[23];
  p.ln2_g = (const float*)d_in[24];
  p.ln2_b = (const float*)d_in[25];
  p.ff1_b = (const float*)d_in[27];
  p.ff2_b = (const float*)d_in[29];
  p.post_g = (const float*)d_in[30];
  p.post_b = (const float*)d_in[31];
  p.at_w1 = (const float*)d_in[32];
  p.at_b1 = (const float*)d_in[33];
  p.at_w2 = (const float*)d_in[34];
  p.at_b2 = (const float*)d_in[35];
  p.hx_w1 = (const float*)d_in[36];
  p.hx_b1 = (const float*)d_in[37];
  p.hx_w2 = (const float*)d_in[38];
  p.hx_b2 = (const float*)d_in[39];
  p.tg_w1 = (const float*)d_in[40];
  p.tg_b1 = (const float*)d_in[41];
  p.tg_w2 = (const float*)d_in[42];
  p.tg_b2 = (const float*)d_in[43];
  p.vl_w1 = (const float*)d_in[44];
  p.vl_b1 = (const float*)d_in[45];
  p.vl_w2 = (const float*)d_in[46];
  p.vl_b2 = (const float*)d_in[47];
  p.wh = (const _Float16*)d_ws;
  p.out = (float*)d_out;

  convert_weights<<<dim3((NLAY * WL_TOT + 255) / 256), dim3(256), 0, stream>>>(
      (const float*)d_in[20], (const float*)d_in[22],
      (const float*)d_in[26], (const float*)d_in[28], (_Float16*)d_ws);

  (void)hipFuncSetAttribute((const void*)battle_fused,
                            hipFuncAttributeMaxDynamicSharedMemorySize, SMEM_BYTES);
  battle_fused<<<dim3(BN), dim3(256), SMEM_BYTES, stream>>>(p);
}
